// GraphConvolution_14190571946025
// MI455X (gfx1250) — matrix operand order repaired and checked by reading, not compiled, not run on hardware
//
#include <hip/hip_runtime.h>

typedef __attribute__((ext_vector_type(16))) _Float16 v16h;
typedef __attribute__((ext_vector_type(8)))  float    v8f;

#define DIM 128
#define LDS_STRIDE 136   // 128 + 8 halves padding to spread LDS banks

// ---------------------------------------------------------------------------
// Kernel 1: zero the accumulator (d_out reused as segment-sum buffer) and cnt
// ---------------------------------------------------------------------------
__global__ void gc_init(float4* __restrict__ acc4, float* __restrict__ cnt,
                        int n_nodes) {
    size_t idx = (size_t)blockIdx.x * blockDim.x + threadIdx.x;
    size_t total4 = (size_t)n_nodes * (DIM / 4);
    if (idx < total4) acc4[idx] = make_float4(0.f, 0.f, 0.f, 0.f);
    if (idx < (size_t)n_nodes) cnt[idx] = 0.0f;
}

// ---------------------------------------------------------------------------
// Kernel 2: convert W (128x128 f32, row-major) to f16 once; L2-resident after
// ---------------------------------------------------------------------------
__global__ void gc_convW(const float* __restrict__ W, _Float16* __restrict__ Wh) {
    int idx = blockIdx.x * blockDim.x + threadIdx.x;
    if (idx < DIM * DIM) Wh[idx] = (_Float16)W[idx];
}

// ---------------------------------------------------------------------------
// Kernel 3: edge scatter. One wave32 per edge: lane l holds float4 of the
// source row, 4 hardware f32 atomics into the destination row. Lane 0 bumps
// the degree counter. h and acc are L2-resident (51 MB each in 192 MB L2).
// ---------------------------------------------------------------------------
__global__ void gc_scatter(const float* __restrict__ h,
                           const int* __restrict__ src,
                           const int* __restrict__ dst,
                           float* __restrict__ acc,
                           float* __restrict__ cnt,
                           int n_edges) {
    int wave = blockIdx.x * (blockDim.x >> 5) + (threadIdx.x >> 5);
    int lane = threadIdx.x & 31;
    if (wave >= n_edges) return;
    int s = src[wave];
    int d = dst[wave];
    const float4* hp = (const float4*)(h + (size_t)s * DIM);
    float4 v = hp[lane];
    float* op = acc + (size_t)d * DIM + lane * 4;
    unsafeAtomicAdd(op + 0, v.x);
    unsafeAtomicAdd(op + 1, v.y);
    unsafeAtomicAdd(op + 2, v.z);
    unsafeAtomicAdd(op + 3, v.w);
    if (lane == 0) unsafeAtomicAdd(&cnt[d], 1.0f);
}

// ---------------------------------------------------------------------------
// Kernel 4: per 16-node tile: agg = cnt>0 ? sum/cnt : h  (f16 into LDS), then
// out = relu(agg @ W^T + b) via v_wmma_f32_16x16x32_f16. 8 waves per block,
// wave w owns output columns [16w, 16w+16). K=128 done as 4 WMMA steps.
// In-place on d_out: all reads happen before the barrier, writes after.
// ---------------------------------------------------------------------------
__global__ void __launch_bounds__(256, 2)
gc_update(const float* __restrict__ h,
          const _Float16* __restrict__ Wh,
          const float* __restrict__ bias,
          const float* __restrict__ cnt,
          float* __restrict__ out,
          int n_nodes) {
    __shared__ _Float16 aggh[16 * LDS_STRIDE];

    const int base = blockIdx.x * 16;
    const int tid  = threadIdx.x;

    // Stage 1: agg row tile -> f16 LDS, float4-vectorized (2 iters/thread)
    for (int i = tid; i < 16 * (DIM / 4); i += 256) {
        int row  = i >> 5;            // 0..15
        int c4   = (i & 31) * 4;      // 0,4,...,124
        int node = base + row;
        float4 val = make_float4(0.f, 0.f, 0.f, 0.f);
        if (node < n_nodes) {
            float deg = cnt[node];
            if (deg > 0.0f) {
                float4 s = *(const float4*)(out + (size_t)node * DIM + c4);
                float inv = 1.0f / deg;
                val = make_float4(s.x * inv, s.y * inv, s.z * inv, s.w * inv);
            } else {
                val = *(const float4*)(h + (size_t)node * DIM + c4);
            }
        }
        union { _Float16 hh[4]; uint2 u; } p;
        p.hh[0] = (_Float16)val.x;
        p.hh[1] = (_Float16)val.y;
        p.hh[2] = (_Float16)val.z;
        p.hh[3] = (_Float16)val.w;
        *(uint2*)(&aggh[row * LDS_STRIDE + c4]) = p.u;
    }
    __syncthreads();

    // Stage 2: WMMA. Fragment layouts per CDNA5 ISA 7.12.2 (wave32).
    const int wave = tid >> 5;          // 0..7 -> output column tile
    const int lane = tid & 31;
    const int hsel = lane >> 4;         // 0: lanes 0-15, 1: lanes 16-31
    const int nlo  = lane & 15;
    const int col_base = wave * 16;

    union Frag { v16h v; uint4 q[2]; };

    v8f c = {};
    const _Float16* arow = &aggh[nlo * LDS_STRIDE];             // A: M = lane&15
    const _Float16* brow = Wh + (size_t)(col_base + nlo) * DIM; // B: N = lane&15

#pragma unroll
    for (int chunk = 0; chunk < 4; ++chunk) {
        const int kc = chunk * 32;
        Frag a, b;
        // A 16x32 f16: lanes 0-15 hold K = kc+{0..7, 16..23},
        //              lanes 16-31 hold K = kc+{8..15, 24..31}
        a.q[0] = *(const uint4*)(arow + kc + hsel * 8);
        a.q[1] = *(const uint4*)(arow + kc + 16 + hsel * 8);
        // B 32x16 f16: same K order as A (lanes 0-15 K = kc+{0..7,16..23}, lanes 16-31 K = kc+{8..15,24..31})
        // B[k][n] = W[col_base+n][k]  (out = agg @ W^T)
        b.q[0] = *(const uint4*)(brow + kc + hsel * 8);
        b.q[1] = *(const uint4*)(brow + kc + 16 + hsel * 8);
        // (neg_a, A, neg_b, B, c_mod, C, reuse_a, reuse_b)
        c = __builtin_amdgcn_wmma_f32_16x16x32_f16(false, a.v, false, b.v,
                                                   (short)0, c, false, false);
    }

    // Epilogue: D layout: VGPR r -> M = r + 8*hsel, N = lane&15.
    // Bounds check is uniform per block -> single scalar branch; full-tile
    // path is 8 unconditional stores off one base with immediate offsets.
    const float bv = bias[col_base + nlo];
    if (base + 16 <= n_nodes) {
        float* orow = out + (size_t)(base + hsel * 8) * DIM + col_base + nlo;
#pragma unroll
        for (int r = 0; r < 8; ++r) {
            float val = c[r] + bv;
            orow[(size_t)r * DIM] = val > 0.0f ? val : 0.0f;
        }
    } else {
#pragma unroll
        for (int r = 0; r < 8; ++r) {
            int node = base + r + hsel * 8;
            float val = c[r] + bv;
            if (node < n_nodes)
                out[(size_t)node * DIM + col_base + nlo] = val > 0.0f ? val : 0.0f;
        }
    }
}

// ---------------------------------------------------------------------------
extern "C" void kernel_launch(void* const* d_in, const int* in_sizes, int n_in,
                              void* d_out, int out_size, void* d_ws, size_t ws_size,
                              hipStream_t stream) {
    const float* h    = (const float*)d_in[0];
    const int*   src  = (const int*)d_in[1];
    const int*   dst  = (const int*)d_in[2];
    const float* W    = (const float*)d_in[3];
    const float* bias = (const float*)d_in[4];
    float*       out  = (float*)d_out;

    const int n_nodes = in_sizes[0] / DIM;
    const int n_edges = in_sizes[1];

    // Workspace layout: [cnt: n_nodes f32][Wh: 128*128 f16]
    float* cnt = (float*)d_ws;
    size_t cnt_bytes = ((size_t)n_nodes * sizeof(float) + 255) & ~(size_t)255;
    _Float16* Wh = (_Float16*)((char*)d_ws + cnt_bytes);

    size_t total4 = (size_t)n_nodes * (DIM / 4);
    int init_blocks = (int)((total4 + 255) / 256);
    gc_init<<<init_blocks, 256, 0, stream>>>((float4*)out, cnt, n_nodes);

    gc_convW<<<(DIM * DIM + 255) / 256, 256, 0, stream>>>(W, Wh);

    int edge_blocks = (n_edges + 7) / 8;   // 8 waves (edges) per 256-thread block
    gc_scatter<<<edge_blocks, 256, 0, stream>>>(h, src, dst, out, cnt, n_edges);

    int node_blocks = (n_nodes + 15) / 16;
    gc_update<<<node_blocks, 256, 0, stream>>>(h, Wh, bias, cnt, out, n_nodes);
}
